// GCN_43207370998446
// MI455X (gfx1250) — compile-verified, matrix operand order checked by reading, not run on hardware
//
#include <hip/hip_runtime.h>
#include <hip/hip_bf16.h>
#include <stdint.h>

// ---------------------------------------------------------------------------
// GNN (PyG GraphConv x4 + mean-pool + MLP) for MI455X (gfx1250, wave32, WMMA)
// bf16-resident activations; weights staged in LDS (padded, conflict-free);
// WMMA K-loop: global b128 loads (A) + ds_load_b128 (B) + v_wmma.
// ---------------------------------------------------------------------------

#define NNODES 100000
#define NEDGES 1600000
#define NGRAPH 512
#define NCLS   10

typedef __attribute__((ext_vector_type(16))) __bf16 v16bf;
typedef __attribute__((ext_vector_type(8)))  float  v8f;

// fp32 -> bf16 round-to-nearest-even (integer path; no __bf16 fp conversions)
__device__ __forceinline__ __bf16 f2bf(float f) {
    unsigned u = __builtin_bit_cast(unsigned, f);
    unsigned r = u + 0x7FFFu + ((u >> 16) & 1u);
    unsigned short h = (unsigned short)(r >> 16);
    __bf16 b;
    __builtin_memcpy(&b, &h, sizeof(b));
    return b;
}

// ---------------------------------------------------------------------------
// Utility kernels
// ---------------------------------------------------------------------------
__global__ void zero_f32(float* __restrict__ p, int nvec4) {
    int i = blockIdx.x * blockDim.x + threadIdx.x;
    if (i < nvec4) reinterpret_cast<float4*>(p)[i] = make_float4(0.f, 0.f, 0.f, 0.f);
}

// bulk fp32 -> bf16, 8 elements / thread (float4 x2 in, uint4 out)
__global__ void cvt8_f32_bf16(const float* __restrict__ s, __bf16* __restrict__ d, int nvec8) {
    int i = blockIdx.x * blockDim.x + threadIdx.x;
    if (i >= nvec8) return;
    const float4* p = reinterpret_cast<const float4*>(s) + 2 * (size_t)i;
    float4 a = p[0], b = p[1];
    union { uint4 q; __bf16 e[8]; } u;
    u.e[0] = f2bf(a.x); u.e[1] = f2bf(a.y); u.e[2] = f2bf(a.z); u.e[3] = f2bf(a.w);
    u.e[4] = f2bf(b.x); u.e[5] = f2bf(b.y); u.e[6] = f2bf(b.z); u.e[7] = f2bf(b.w);
    reinterpret_cast<uint4*>(d)[i] = u.q;
}

// ---------------------------------------------------------------------------
// Edge scatter-add: agg[dst] += h[src]   (fp32, L2-resident atomics)
// ---------------------------------------------------------------------------
template <int FIN>
__global__ void scatter_add(const float* __restrict__ h,
                            const int* __restrict__ src,
                            const int* __restrict__ dst,
                            float* __restrict__ agg, int nE) {
    const int c = threadIdx.x;                       // float4 chunk
    const int e = blockIdx.x * blockDim.y + threadIdx.y;
    if (e >= nE) return;
    const int s = __ldg(src + e);
    const int d = __ldg(dst + e);
    const float4 v = *reinterpret_cast<const float4*>(h + (size_t)s * FIN + 4 * c);
    float* o = agg + (size_t)d * FIN + 4 * c;
    atomicAdd(o + 0, v.x);
    atomicAdd(o + 1, v.y);
    atomicAdd(o + 2, v.z);
    atomicAdd(o + 3, v.w);
}

// ---------------------------------------------------------------------------
// WMMA 16x32 bf16 fragments: per CDNA5 ISA §7.12.2 each lane's K
// values form two contiguous 8-element runs -> two b128 loads (global or LDS).
// ---------------------------------------------------------------------------
__device__ __forceinline__ v16bf load_frag16_g(const __bf16* __restrict__ base) {
    union { v16bf v; uint4 q[2]; } u;
    const uint4* p = reinterpret_cast<const uint4*>(base);
    u.q[0] = p[0];   // K kb..kb+7
    u.q[1] = p[2];   // K kb+16..kb+23  (+16 elems = +32B)
    return u.v;
}

__device__ __forceinline__ v16bf load_frag16_lds(const __bf16* base) {
    union { v16bf v; uint4 q[2]; } u;
    const uint4* p = reinterpret_cast<const uint4*>(base);
    u.q[0] = p[0];
    u.q[1] = p[2];
    return u.v;
}

// ---------------------------------------------------------------------------
// Fused GraphConv GEMM: out = agg @ Wr^T + h @ Ws^T + bias (+ReLU)
// Weights staged in LDS once per block (row pitch K+8 -> conflict-free
// ds_load_b128: column stride advances 4 banks, 16 lanes tile 64 banks).
// One wave computes a 16x64 output strip: 8 x v_wmma per K-step, A reused
// across the 4 n-tiles. Emits fp32 output + optional bf16 shadow.
// ---------------------------------------------------------------------------
template <int K, int FOUT, bool RELU, bool EMIT_BF>
__global__ __launch_bounds__(256)
void gconv_gemm(const __bf16* __restrict__ aggb, const __bf16* __restrict__ hb,
                const __bf16* __restrict__ Wr, const __bf16* __restrict__ Ws,
                const float* __restrict__ bias, float* __restrict__ outf,
                __bf16* __restrict__ outb, int mtiles) {
    constexpr int NSTRIP = 4;             // n-tiles per wave (16x64 strip)
    constexpr int WN = FOUT / 64;         // wave columns per row band
    constexpr int PITCH = K + 8;          // LDS row pitch (bank-decorrelated)

    __shared__ __align__(16) __bf16 sW[2 * FOUT * PITCH];
    __bf16* sWr = sW;
    __bf16* sWs = sW + FOUT * PITCH;

    // Cooperative stage of both weight matrices into LDS (uint4 granularity)
    {
        constexpr int CH = K / 8;         // 8-element chunks per row
        for (int i = threadIdx.x; i < FOUT * CH; i += 256) {
            const int r = i / CH, c = i % CH;
            reinterpret_cast<uint4*>(sWr + r * PITCH)[c] =
                reinterpret_cast<const uint4*>(Wr + (size_t)r * K)[c];
            reinterpret_cast<uint4*>(sWs + r * PITCH)[c] =
                reinterpret_cast<const uint4*>(Ws + (size_t)r * K)[c];
        }
    }
    __syncthreads();   // must precede the wave guard (partial last block)

    const int wave = (blockIdx.x * blockDim.x + threadIdx.x) >> 5;
    const int lane = threadIdx.x & 31;
    if (wave >= mtiles * WN) return;      // uniform per wave: EXEC stays all-1s

    const int mt   = wave / WN;
    const int n0   = (wave % WN) * 64;
    const int rsel = lane & 15;
    const int kb   = (lane >> 4) * 8;     // 0 or 8
    const int row  = mt * 16 + rsel;

    v8f acc[NSTRIP] = {{}, {}, {}, {}};
    #pragma unroll
    for (int k0 = 0; k0 < K; k0 += 32) {
        const v16bf fa = load_frag16_g(aggb + (size_t)row * K + k0 + kb);
        const v16bf fh = load_frag16_g(hb   + (size_t)row * K + k0 + kb);
        #pragma unroll
        for (int t = 0; t < NSTRIP; ++t) {
            const int col = n0 + t * 16 + rsel;
            const v16bf br = load_frag16_lds(sWr + col * PITCH + k0 + kb);
            const v16bf bs = load_frag16_lds(sWs + col * PITCH + k0 + kb);
            acc[t] = __builtin_amdgcn_wmma_f32_16x16x32_bf16(false, fa, false, br,
                                                             (short)0, acc[t], false, false);
            acc[t] = __builtin_amdgcn_wmma_f32_16x16x32_bf16(false, fh, false, bs,
                                                             (short)0, acc[t], false, false);
        }
    }

    #pragma unroll
    for (int t = 0; t < NSTRIP; ++t) {
        const int col = n0 + t * 16 + rsel;
        const float bc = bias[col];
        #pragma unroll
        for (int r = 0; r < 8; ++r) {
            const int m = mt * 16 + ((lane < 16) ? r : r + 8);
            float v = acc[t][r] + bc;
            if (RELU) v = fmaxf(v, 0.f);
            outf[(size_t)m * FOUT + col] = v;
            if (EMIT_BF) outb[(size_t)m * FOUT + col] = f2bf(v);
        }
    }
}

// ---------------------------------------------------------------------------
// global_mean_pool: sums[batch[i]] += h[i], cnt[batch[i]] += 1; then divide
// ---------------------------------------------------------------------------
__global__ void pool_accum(const float* __restrict__ h, const int* __restrict__ batch,
                           float* __restrict__ sums, float* __restrict__ cnt, int n) {
    const int c = threadIdx.x;                       // 0..31 (float4 chunks of 128)
    const int i = blockIdx.x * blockDim.y + threadIdx.y;
    if (i >= n) return;
    const int b = __ldg(batch + i);
    const float4 v = *reinterpret_cast<const float4*>(h + (size_t)i * 128 + 4 * c);
    float* o = sums + (size_t)b * 128 + 4 * c;
    atomicAdd(o + 0, v.x);
    atomicAdd(o + 1, v.y);
    atomicAdd(o + 2, v.z);
    atomicAdd(o + 3, v.w);
    if (c == 0) atomicAdd(cnt + b, 1.0f);
}

__global__ void pool_div(float* __restrict__ sums, const float* __restrict__ cnt, int n) {
    int i = blockIdx.x * blockDim.x + threadIdx.x;
    if (i < n) sums[i] /= fmaxf(cnt[i >> 7], 1.0f);
}

// ---------------------------------------------------------------------------
// Small dense linear (head): out[M,Fo] = in[M,Kd] @ W[Fo,Kd]^T + b
// ---------------------------------------------------------------------------
__global__ void linear_small(const float* __restrict__ in, const float* __restrict__ W,
                             const float* __restrict__ b, float* __restrict__ out,
                             int M, int Kd, int Fo) {
    int i = blockIdx.x * blockDim.x + threadIdx.x;
    if (i >= M * Fo) return;
    const int m = i / Fo, f = i % Fo;
    const float* xr = in + (size_t)m * Kd;
    const float* wr = W + (size_t)f * Kd;
    float s = b[f];
    for (int k = 0; k < Kd; ++k) s = fmaf(xr[k], wr[k], s);
    out[i] = s;
}

// ---------------------------------------------------------------------------
// Host orchestration (capture-safe: everything on `stream`, scratch in d_ws)
// ---------------------------------------------------------------------------
extern "C" void kernel_launch(void* const* d_in, const int* in_sizes, int n_in,
                              void* d_out, int out_size, void* d_ws, size_t ws_size,
                              hipStream_t stream) {
    (void)in_sizes; (void)n_in; (void)out_size; (void)ws_size;

    const float* x     = (const float*)d_in[0];
    const int*   ei    = (const int*)d_in[1];
    const int*   src   = ei;              // edge_index[0]
    const int*   dst   = ei + NEDGES;     // edge_index[1]
    const int*   batch = (const int*)d_in[2];
    const float* w1r = (const float*)d_in[3],  *w1s = (const float*)d_in[4],  *b1 = (const float*)d_in[5];
    const float* w2r = (const float*)d_in[6],  *w2s = (const float*)d_in[7],  *b2 = (const float*)d_in[8];
    const float* w3r = (const float*)d_in[9],  *w3s = (const float*)d_in[10], *b3 = (const float*)d_in[11];
    const float* w4r = (const float*)d_in[12], *w4s = (const float*)d_in[13], *b4 = (const float*)d_in[14];
    const float* wl1 = (const float*)d_in[15], *bl1 = (const float*)d_in[16];
    const float* wl2 = (const float*)d_in[17], *bl2 = (const float*)d_in[18];
    float* out = (float*)d_out;

    // Workspace layout: fp32 region then bf16 region
    float* ws   = (float*)d_ws;
    float* hAf  = ws;                                  // N*128 f32
    float* hBf  = hAf + (size_t)NNODES * 128;          // N*128 f32
    float* aggf = hBf + (size_t)NNODES * 128;          // N*128 f32
    float* sums = aggf + (size_t)NNODES * 128;         // 512*128
    float* cnt  = sums + (size_t)NGRAPH * 128;         // 512
    float* g1   = cnt + NGRAPH;                        // 512*64
    __bf16* bf   = (__bf16*)(g1 + (size_t)NGRAPH * 64);
    __bf16* hAb  = bf;                                 // N*128 bf16
    __bf16* hBb  = hAb + (size_t)NNODES * 128;         // N*128 bf16
    __bf16* xb   = hBb;                                // alias: x_bf16 lives in hBb until L2
    __bf16* aggb = hBb + (size_t)NNODES * 128;         // N*128 bf16
    __bf16* wbf  = aggb + (size_t)NNODES * 128;        // 90112 bf16 weights
    __bf16 *w1rb = wbf,          *w1sb = wbf + 4096;
    __bf16 *w2rb = wbf + 8192,   *w2sb = wbf + 16384;
    __bf16 *w3rb = wbf + 24576,  *w3sb = wbf + 40960;
    __bf16 *w4rb = wbf + 57344,  *w4sb = wbf + 73728;

    // --- Convert weights + x to bf16 (row-major [fout][fin] = WMMA B operand)
    cvt8_f32_bf16<<<(4096  / 8 + 255) / 256, 256, 0, stream>>>(w1r, w1rb, 4096 / 8);
    cvt8_f32_bf16<<<(4096  / 8 + 255) / 256, 256, 0, stream>>>(w1s, w1sb, 4096 / 8);
    cvt8_f32_bf16<<<(8192  / 8 + 255) / 256, 256, 0, stream>>>(w2r, w2rb, 8192 / 8);
    cvt8_f32_bf16<<<(8192  / 8 + 255) / 256, 256, 0, stream>>>(w2s, w2sb, 8192 / 8);
    cvt8_f32_bf16<<<(16384 / 8 + 255) / 256, 256, 0, stream>>>(w3r, w3rb, 16384 / 8);
    cvt8_f32_bf16<<<(16384 / 8 + 255) / 256, 256, 0, stream>>>(w3s, w3sb, 16384 / 8);
    cvt8_f32_bf16<<<(16384 / 8 + 255) / 256, 256, 0, stream>>>(w4r, w4rb, 16384 / 8);
    cvt8_f32_bf16<<<(16384 / 8 + 255) / 256, 256, 0, stream>>>(w4s, w4sb, 16384 / 8);
    cvt8_f32_bf16<<<((NNODES * 64 / 8) + 255) / 256, 256, 0, stream>>>(x, xb, NNODES * 64 / 8);

    const int MT = NNODES / 16;                    // 6250 exact
    const int gvb64  = (MT * 1 * 32 + 255) / 256;  // FOUT=64  -> 6250 waves
    const int gvb128 = (MT * 2 * 32 + 255) / 256;  // FOUT=128 -> 12500 waves
    const dim3 sblk64(16, 16), sblk128(32, 8);
    const int sg64  = (NEDGES + 15) / 16;
    const int sg128 = (NEDGES + 7) / 8;

    // --- Layer 1: fin=64, fout=64, relu.  h = x (fp32 in d_in, bf16 in xb)
    zero_f32<<<((NNODES * 64 / 4) + 255) / 256, 256, 0, stream>>>(aggf, NNODES * 64 / 4);
    scatter_add<64><<<sg64, sblk64, 0, stream>>>(x, src, dst, aggf, NEDGES);
    cvt8_f32_bf16<<<((NNODES * 64 / 8) + 255) / 256, 256, 0, stream>>>(aggf, aggb, NNODES * 64 / 8);
    gconv_gemm<64, 64, true, true><<<gvb64, 256, 0, stream>>>(aggb, xb, w1rb, w1sb, b1, hAf, hAb, MT);

    // --- Layer 2: fin=64, fout=128, relu  (overwrites xb alias region in hBb)
    zero_f32<<<((NNODES * 64 / 4) + 255) / 256, 256, 0, stream>>>(aggf, NNODES * 64 / 4);
    scatter_add<64><<<sg64, sblk64, 0, stream>>>(hAf, src, dst, aggf, NEDGES);
    cvt8_f32_bf16<<<((NNODES * 64 / 8) + 255) / 256, 256, 0, stream>>>(aggf, aggb, NNODES * 64 / 8);
    gconv_gemm<64, 128, true, true><<<gvb128, 256, 0, stream>>>(aggb, hAb, w2rb, w2sb, b2, hBf, hBb, MT);

    // --- Layer 3: fin=128, fout=128, relu
    zero_f32<<<((NNODES * 128 / 4) + 255) / 256, 256, 0, stream>>>(aggf, NNODES * 128 / 4);
    scatter_add<128><<<sg128, sblk128, 0, stream>>>(hBf, src, dst, aggf, NEDGES);
    cvt8_f32_bf16<<<((NNODES * 128 / 8) + 255) / 256, 256, 0, stream>>>(aggf, aggb, NNODES * 128 / 8);
    gconv_gemm<128, 128, true, true><<<gvb128, 256, 0, stream>>>(aggb, hBb, w3rb, w3sb, b3, hAf, hAb, MT);

    // --- Layer 4: fin=128, fout=128, no relu, no bf16 shadow needed
    zero_f32<<<((NNODES * 128 / 4) + 255) / 256, 256, 0, stream>>>(aggf, NNODES * 128 / 4);
    scatter_add<128><<<sg128, sblk128, 0, stream>>>(hAf, src, dst, aggf, NEDGES);
    cvt8_f32_bf16<<<((NNODES * 128 / 8) + 255) / 256, 256, 0, stream>>>(aggf, aggb, NNODES * 128 / 8);
    gconv_gemm<128, 128, false, false><<<gvb128, 256, 0, stream>>>(aggb, hAb, w4rb, w4sb, b4, hBf, nullptr, MT);

    // --- global_mean_pool (sums and cnt contiguous -> one zero pass)
    const int poolZ = (NGRAPH * 128 + NGRAPH) / 4;
    zero_f32<<<(poolZ + 255) / 256, 256, 0, stream>>>(sums, poolZ);
    pool_accum<<<(NNODES + 7) / 8, dim3(32, 8), 0, stream>>>(hBf, batch, sums, cnt, NNODES);
    pool_div<<<(NGRAPH * 128 + 255) / 256, 256, 0, stream>>>(sums, cnt, NGRAPH * 128);

    // --- MLP head: [512,128] -> [512,64] -> [512,10]
    linear_small<<<(NGRAPH * 64 + 255) / 256, 256, 0, stream>>>(sums, wl1, bl1, g1,
                                                                NGRAPH, 128, 64);
    linear_small<<<(NGRAPH * NCLS + 255) / 256, 256, 0, stream>>>(g1, wl2, bl2, out,
                                                                  NGRAPH, 64, NCLS);
}
